// GlobalFeatureGAT_32693291057325
// MI455X (gfx1250) — compile-verified, matrix operand order checked by reading, not run on hardware
//
#include <hip/hip_runtime.h>
#include <hip/hip_bf16.h>
#include <stdint.h>

typedef __attribute__((ext_vector_type(16))) _Float16 v16h;
typedef __attribute__((ext_vector_type(8)))  _Float16 v8h;
typedef __attribute__((ext_vector_type(8)))  float    v8f;

// ---------------------------------------------------------------------------
// helpers
// ---------------------------------------------------------------------------
__device__ __forceinline__ unsigned fkey(float f) {
    unsigned u = __float_as_uint(f);
    return (u & 0x80000000u) ? ~u : (u | 0x80000000u);   // monotonic uint key
}
__device__ __forceinline__ float fkeyinv(unsigned k) {
    unsigned u = (k & 0x80000000u) ? (k & 0x7FFFFFFFu) : ~k;
    return __uint_as_float(u);
}

// ---------------------------------------------------------------------------
// zero-init (floats and uint keys share bit pattern 0)
// ---------------------------------------------------------------------------
__global__ void zero_kernel(unsigned* p, long long n) {
    long long stride = (long long)gridDim.x * blockDim.x;
    for (long long i = (long long)blockIdx.x * blockDim.x + threadIdx.x; i < n; i += stride)
        p[i] = 0u;
}

// ---------------------------------------------------------------------------
// node features: 5 masked-mean embeddings + scalar concat + layernorm -> f16
// one block (256 threads) per node; each thread owns one feature column
// ---------------------------------------------------------------------------
__global__ void feat_ln_kernel(const float* __restrict__ xs,
                               const int* __restrict__ xop, const int* __restrict__ xsrc,
                               const int* __restrict__ xsink, const int* __restrict__ xstr,
                               const int* __restrict__ xpay,
                               const float* __restrict__ Eop, const float* __restrict__ Esrc,
                               const float* __restrict__ Esink, const float* __restrict__ Estr,
                               const float* __restrict__ Epay,
                               const float* __restrict__ g, const float* __restrict__ b,
                               _Float16* __restrict__ feats) {
    int n = blockIdx.x;
    int t = threadIdx.x;
    float v;
    if (t < 96) {
        v = xs[(size_t)n * 96 + t];
    } else {
        int tb = (t - 96) >> 5, dim = (t - 96) & 31;
        const int* idx; const float* tab; int L;
        switch (tb) {
            case 0:  idx = xop  + (size_t)n * 16; tab = Eop;  L = 16; break;
            case 1:  idx = xsrc + (size_t)n * 8;  tab = Esrc; L = 8;  break;
            case 2:  idx = xsink+ (size_t)n * 8;  tab = Esink;L = 8;  break;
            case 3:  idx = xstr + (size_t)n * 8;  tab = Estr; L = 8;  break;
            default: idx = xpay + (size_t)n * 8;  tab = Epay; L = 8;  break;
        }
        float s = 0.f, c = 0.f;
        for (int i = 0; i < L; ++i) {
            int id = idx[i];
            if (id != 0) { s += tab[(size_t)id * 32 + dim]; c += 1.f; }
        }
        v = s / (c + 1e-9f);
    }
    __shared__ float red[256];
    red[t] = v; __syncthreads();
    for (int off = 128; off > 0; off >>= 1) {
        if (t < off) red[t] += red[t + off];
        __syncthreads();
    }
    float mu = red[0] * (1.f / 256.f);
    __syncthreads();
    float d = v - mu;
    red[t] = d * d; __syncthreads();
    for (int off = 128; off > 0; off >>= 1) {
        if (t < off) red[t] += red[t + off];
        __syncthreads();
    }
    float var = red[0] * (1.f / 256.f);
    float o = d * rsqrtf(var + 1e-5f) * g[t] + b[t];
    feats[(size_t)n * 256 + t] = (_Float16)o;
}

// ---------------------------------------------------------------------------
// pack W [K,NC] row-major f32 -> WMMA B-operand register order, f16
// layout: per (ntile,kstep) tile: 32 lanes x 16 halves contiguous
// ---------------------------------------------------------------------------
__global__ void pack_b_kernel(const float* __restrict__ W, _Float16* __restrict__ P,
                              int K, int NC) {
    int idx = blockIdx.x * blockDim.x + threadIdx.x;
    int total = K * NC;
    if (idx >= total) return;
    int hw   = idx & 15;
    int lane = (idx >> 4) & 31;
    int tile = idx >> 9;
    int ksteps = K >> 5;
    int ntile = tile / ksteps;
    int kstep = tile % ksteps;
    int j = hw >> 1, h = hw & 1;
    int kk = (j < 4) ? (j * 2 + h) : (16 + (j - 4) * 2 + h);
    kk += (lane < 16) ? 0 : 8;
    int n = ntile * 16 + (lane & 15);
    int k = kstep * 32 + kk;
    P[idx] = (_Float16)W[(size_t)k * NC + n];
}

// ---------------------------------------------------------------------------
// WMMA GEMM: D[M,NC](f16) = A[M,K](f16) @ Bpacked ; one wave -> 16x64 tile
// ---------------------------------------------------------------------------
__global__ void wmma_gemm_kernel(const _Float16* __restrict__ A,
                                 const _Float16* __restrict__ Bp,
                                 _Float16* __restrict__ Dst,
                                 int M, int K, int NC, int totalWaves) {
    int wave = (blockIdx.x * blockDim.x + threadIdx.x) >> 5;
    int lane = threadIdx.x & 31;
    if (wave >= totalWaves) return;          // wave-uniform: EXEC stays all-ones
    int groups = NC >> 6;                    // 64-col groups
    int mtile = wave / groups;
    int ngrp  = wave % groups;
    int ksteps = K >> 5;

    int row  = mtile * 16 + (lane & 15);
    int koff = (lane < 16) ? 0 : 8;
    const v8h* Arow = (const v8h*)(A + (size_t)row * K);

    v8f acc[4] = {v8f{}, v8f{}, v8f{}, v8f{}};

    for (int ks = 0; ks < ksteps; ++ks) {
        int kb = ks * 32;
        v8h c0 = Arow[(kb + koff) >> 3];        // K = kb+koff .. +7  -> vgprs 0..3
        v8h c1 = Arow[(kb + 16 + koff) >> 3];   // K = kb+16+koff ..  -> vgprs 4..7
        v16h a;
#pragma unroll
        for (int i = 0; i < 8; ++i) { a[i] = c0[i]; a[i + 8] = c1[i]; }
#pragma unroll
        for (int t = 0; t < 4; ++t) {
            int ntile = ngrp * 4 + t;
            const v8h* bp = (const v8h*)(Bp + (((size_t)ntile * ksteps + ks) * 32 + lane) * 16);
            v8h b0 = bp[0], b1 = bp[1];
            v16h bb;
#pragma unroll
            for (int i = 0; i < 8; ++i) { bb[i] = b0[i]; bb[i + 8] = b1[i]; }
            acc[t] = __builtin_amdgcn_wmma_f32_16x16x32_f16(
                false, a, false, bb, (short)0, acc[t], false, false);
        }
    }
    // D layout: vgpr j -> row mtile*16 + (lane<16 ? j : j+8), col = 16*ntile + (lane&15)
    int rbase = mtile * 16 + ((lane < 16) ? 0 : 8);
    int cl = lane & 15;
#pragma unroll
    for (int t = 0; t < 4; ++t) {
        int col = (ngrp * 4 + t) * 16 + cl;
#pragma unroll
        for (int j = 0; j < 8; ++j) {
            int r = rbase + j;
            if (r < M) Dst[(size_t)r * NC + col] = (_Float16)acc[t][j];
        }
    }
}

// ---------------------------------------------------------------------------
// per-node attention dots: a_src[n,h] = sum_c h[n,h,c]*att_src[h,c] (flat HC)
// blockDim.x == HC, one node per block
// ---------------------------------------------------------------------------
__global__ void attn_vec_kernel(const _Float16* __restrict__ h,
                                const float* __restrict__ att_s,
                                const float* __restrict__ att_d,
                                float* __restrict__ as_out, float* __restrict__ ad_out,
                                int HC, int H, int Cc) {
    extern __shared__ float sm[];            // 2*HC
    int n = blockIdx.x, t = threadIdx.x;
    float v = (float)h[(size_t)n * HC + t];
    sm[t]      = v * att_s[t];
    sm[HC + t] = v * att_d[t];
    __syncthreads();
    int loc = t % Cc;
    for (int off = Cc >> 1; off > 0; off >>= 1) {
        if (loc < off) { sm[t] += sm[t + off]; sm[HC + t] += sm[HC + t + off]; }
        __syncthreads();
    }
    if (loc == 0) {
        int grp = t / Cc;
        as_out[(size_t)n * H + grp] = sm[t];
        ad_out[(size_t)n * H + grp] = sm[HC + t];
    }
}

// ---------------------------------------------------------------------------
// edge pass 1: segment max of leaky_relu(a_src[s]+a_dst[d]) via key atomicMax
// ---------------------------------------------------------------------------
__global__ void edge_max_kernel(const int* __restrict__ src, const int* __restrict__ dst,
                                int E, long long total,
                                const float* __restrict__ as, const float* __restrict__ ad,
                                unsigned* __restrict__ mkey, int H) {
    long long stride = (long long)gridDim.x * blockDim.x;
    for (long long idx = (long long)blockIdx.x * blockDim.x + threadIdx.x; idx < total; idx += stride) {
        int hh = (int)(idx % H);
        long long e = idx / H;
        int s = (e < E) ? src[e] : (int)(e - E);
        int d = (e < E) ? dst[e] : (int)(e - E);
        float v = as[(size_t)s * H + hh] + ad[(size_t)d * H + hh];
        v = v > 0.f ? v : 0.2f * v;
        atomicMax(&mkey[(size_t)d * H + hh], fkey(v));
    }
}

// ---------------------------------------------------------------------------
// edge pass 2: ex = exp(v - max); store per-edge; atomicAdd denom
// ---------------------------------------------------------------------------
__global__ void edge_exp_kernel(const int* __restrict__ src, const int* __restrict__ dst,
                                int E, long long total,
                                const float* __restrict__ as, const float* __restrict__ ad,
                                const unsigned* __restrict__ mkey,
                                float* __restrict__ exv, float* __restrict__ den, int H) {
    long long stride = (long long)gridDim.x * blockDim.x;
    for (long long idx = (long long)blockIdx.x * blockDim.x + threadIdx.x; idx < total; idx += stride) {
        int hh = (int)(idx % H);
        long long e = idx / H;
        int s = (e < E) ? src[e] : (int)(e - E);
        int d = (e < E) ? dst[e] : (int)(e - E);
        float v = as[(size_t)s * H + hh] + ad[(size_t)d * H + hh];
        v = v > 0.f ? v : 0.2f * v;
        float m = fkeyinv(mkey[(size_t)d * H + hh]);
        float ex = __expf(v - m);
        exv[idx] = ex;
        atomicAdd(&den[(size_t)d * H + hh], ex);
    }
}

// ---------------------------------------------------------------------------
// edge pass 3: out[d, col] += h[s, col] * (ex / (den+1e-16))
// ---------------------------------------------------------------------------
__global__ void edge_msg_kernel(const int* __restrict__ src, const int* __restrict__ dst,
                                int E, long long total,
                                const _Float16* __restrict__ h,
                                const float* __restrict__ exv, const float* __restrict__ den,
                                float* __restrict__ out, int H, int HC) {
    long long stride = (long long)gridDim.x * blockDim.x;
    int Cc = HC / H;
    for (long long idx = (long long)blockIdx.x * blockDim.x + threadIdx.x; idx < total; idx += stride) {
        int col = (int)(idx % HC);
        long long e = idx / HC;
        int s = (e < E) ? src[e] : (int)(e - E);
        int d = (e < E) ? dst[e] : (int)(e - E);
        int hh = col / Cc;
        float alpha = exv[(size_t)e * H + hh] / (den[(size_t)d * H + hh] + 1e-16f);
        atomicAdd(&out[(size_t)d * HC + col], (float)h[(size_t)s * HC + col] * alpha);
    }
}

// ---------------------------------------------------------------------------
// bias + ELU, f16 or f32 out
// ---------------------------------------------------------------------------
__global__ void bias_elu_f16_kernel(const float* __restrict__ in, const float* __restrict__ bias,
                                    _Float16* __restrict__ out, long long total, int HC) {
    long long stride = (long long)gridDim.x * blockDim.x;
    for (long long i = (long long)blockIdx.x * blockDim.x + threadIdx.x; i < total; i += stride) {
        float v = in[i] + bias[i % HC];
        v = v > 0.f ? v : expm1f(v);
        out[i] = (_Float16)v;
    }
}
__global__ void bias_elu_f32_kernel(const float* __restrict__ in, const float* __restrict__ bias,
                                    float* __restrict__ out, long long total, int HC) {
    long long stride = (long long)gridDim.x * blockDim.x;
    for (long long i = (long long)blockIdx.x * blockDim.x + threadIdx.x; i < total; i += stride) {
        float v = in[i] + bias[i % HC];
        v = v > 0.f ? v : expm1f(v);
        out[i] = v;
    }
}

// ---------------------------------------------------------------------------
// global max pool per graph
// ---------------------------------------------------------------------------
__global__ void pool_max_kernel(const float* __restrict__ x, const int* __restrict__ batch,
                                unsigned* __restrict__ pk, long long total, int Cc) {
    long long stride = (long long)gridDim.x * blockDim.x;
    for (long long i = (long long)blockIdx.x * blockDim.x + threadIdx.x; i < total; i += stride) {
        int n = (int)(i / Cc), c = (int)(i % Cc);
        atomicMax(&pk[(size_t)batch[n] * Cc + c], fkey(x[i]));
    }
}

// ---------------------------------------------------------------------------
// classifier
// ---------------------------------------------------------------------------
__global__ void cls1_kernel(const unsigned* __restrict__ pk, const float* __restrict__ Wc1,
                            const float* __restrict__ bc1, float* __restrict__ hcls) {
    int g = blockIdx.x, j = threadIdx.x;            // 64 blocks x 64 threads
    float s = bc1[j];
    for (int k = 0; k < 128; ++k)
        s += fkeyinv(pk[g * 128 + k]) * Wc1[k * 64 + j];
    hcls[g * 64 + j] = fmaxf(s, 0.f);
}
__global__ void cls2_kernel(const float* __restrict__ hcls, const float* __restrict__ Wc2,
                            const float* __restrict__ bc2, float* __restrict__ out) {
    int g = blockIdx.x, j = threadIdx.x;            // 64 blocks x 8 threads (5 used)
    if (j >= 5) return;
    float s = bc2[j];
    for (int k = 0; k < 64; ++k)
        s += hcls[g * 64 + k] * Wc2[k * 5 + j];
    out[g * 5 + j] = s;
}

// ---------------------------------------------------------------------------
// host launcher
// ---------------------------------------------------------------------------
static inline int gsgrid(long long n) {
    long long b = (n + 255) / 256;
    if (b > (1 << 20)) b = (1 << 20);
    return (int)b;
}

extern "C" void kernel_launch(void* const* d_in, const int* in_sizes, int n_in,
                              void* d_out, int out_size, void* d_ws, size_t ws_size,
                              hipStream_t stream) {
    const int N  = in_sizes[0] / 96;
    const int E  = in_sizes[6] / 2;
    const int Et = E + N;

    const float* xs    = (const float*)d_in[0];
    const int*   xop   = (const int*)d_in[1];
    const int*   xsrc  = (const int*)d_in[2];
    const int*   xsink = (const int*)d_in[3];
    const int*   xstr  = (const int*)d_in[4];
    const int*   xpay  = (const int*)d_in[5];
    const int*   eidx  = (const int*)d_in[6];
    const int*   batch = (const int*)d_in[7];
    const float* Eop   = (const float*)d_in[8];
    const float* Esrc  = (const float*)d_in[9];
    const float* Esink = (const float*)d_in[10];
    const float* Estr  = (const float*)d_in[11];
    const float* Epay  = (const float*)d_in[12];
    const float* ln_g  = (const float*)d_in[13];
    const float* ln_b  = (const float*)d_in[14];
    const float* W1    = (const float*)d_in[15];
    const float* att_s1= (const float*)d_in[16];
    const float* att_d1= (const float*)d_in[17];
    const float* b1    = (const float*)d_in[18];
    const float* W2    = (const float*)d_in[19];
    const float* att_s2= (const float*)d_in[20];
    const float* att_d2= (const float*)d_in[21];
    const float* b2    = (const float*)d_in[22];
    const float* Wc1   = (const float*)d_in[23];
    const float* bc1   = (const float*)d_in[24];
    const float* Wc2   = (const float*)d_in[25];
    const float* bc2   = (const float*)d_in[26];

    const int* src = eidx;          // edge_index[0]
    const int* dst = eidx + E;      // edge_index[1]

    // ---- workspace layout (256B aligned) with explicit reuse ----
    char* base = (char*)d_ws;
    size_t off = 0;
    auto alloc = [&](size_t bytes) -> char* {
        char* p = base + off;
        off = (off + bytes + 255) & ~(size_t)255;
        return p;
    };
    _Float16* feats = (_Float16*)alloc((size_t)N * 256 * 2);   // also reused for h2
    _Float16* W1p   = (_Float16*)alloc(256 * 512 * 2);
    _Float16* W2p   = (_Float16*)alloc(512 * 128 * 2);
    _Float16* h1    = (_Float16*)alloc((size_t)N * 512 * 2);   // reused for out2/x2
    float*    as1   = (float*)alloc((size_t)N * 4 * 4);        // reused for as2
    float*    ad1   = (float*)alloc((size_t)N * 4 * 4);        // reused for ad2
    unsigned* mk1   = (unsigned*)alloc((size_t)N * 4 * 4);     // reused for mk2
    float*    den1  = (float*)alloc((size_t)N * 4 * 4);        // reused for den2
    float*    ex1   = (float*)alloc((size_t)Et * 4 * 4);       // reused for ex2
    float*    out1  = (float*)alloc((size_t)N * 512 * 4);
    _Float16* x1    = (_Float16*)alloc((size_t)N * 512 * 2);
    unsigned* pk    = (unsigned*)alloc((size_t)64 * 128 * 4);
    float*    hcls  = (float*)alloc((size_t)64 * 64 * 4);
    // aliases (lifetimes disjoint from originals)
    _Float16* h2   = feats;                  // [N,128] f16 (12.8MB <= 25.6MB)
    float*    out2 = (float*)h1;             // [N,128] f32 (25.6MB <= 51.2MB)
    float*    x2   = (float*)((char*)h1 + (size_t)N * 128 * 4); // next 25.6MB of h1 region
    float*    as2 = as1; float* ad2 = ad1; unsigned* mk2 = mk1; float* den2 = den1;
    float*    ex2 = ex1;

    // ---- zero-init conv1 accumulators + pool keys ----
    zero_kernel<<<gsgrid((long long)N * 4), 256, 0, stream>>>(mk1, (long long)N * 4);
    zero_kernel<<<gsgrid((long long)N * 4), 256, 0, stream>>>((unsigned*)den1, (long long)N * 4);
    zero_kernel<<<gsgrid((long long)N * 512), 256, 0, stream>>>((unsigned*)out1, (long long)N * 512);
    zero_kernel<<<1, 256, 0, stream>>>(pk, 64 * 128);

    // ---- node features + layernorm -> f16 ----
    feat_ln_kernel<<<N, 256, 0, stream>>>(xs, xop, xsrc, xsink, xstr, xpay,
                                          Eop, Esrc, Esink, Estr, Epay,
                                          ln_g, ln_b, feats);

    // ---- pack weights into WMMA-B order ----
    pack_b_kernel<<<(256 * 512 + 255) / 256, 256, 0, stream>>>(W1, W1p, 256, 512);
    pack_b_kernel<<<(512 * 128 + 255) / 256, 256, 0, stream>>>(W2, W2p, 512, 128);

    // ---- GEMM1: h1 = feats @ W1   [N,256]x[256,512] ----
    {
        int totalWaves = ((N + 15) / 16) * (512 / 64);
        int blocks = (totalWaves * 32 + 255) / 256;
        wmma_gemm_kernel<<<blocks, 256, 0, stream>>>(feats, W1p, h1, N, 256, 512, totalWaves);
    }

    // ---- conv1 attention ----
    attn_vec_kernel<<<N, 512, 2 * 512 * 4, stream>>>(h1, att_s1, att_d1, as1, ad1, 512, 4, 128);
    edge_max_kernel<<<gsgrid((long long)Et * 4), 256, 0, stream>>>(src, dst, E, (long long)Et * 4,
                                                                   as1, ad1, mk1, 4);
    edge_exp_kernel<<<gsgrid((long long)Et * 4), 256, 0, stream>>>(src, dst, E, (long long)Et * 4,
                                                                   as1, ad1, mk1, ex1, den1, 4);
    edge_msg_kernel<<<gsgrid((long long)Et * 512), 256, 0, stream>>>(src, dst, E, (long long)Et * 512,
                                                                     h1, ex1, den1, out1, 4, 512);
    bias_elu_f16_kernel<<<gsgrid((long long)N * 512), 256, 0, stream>>>(out1, b1, x1,
                                                                        (long long)N * 512, 512);

    // ---- GEMM2: h2 = x1 @ W2   [N,512]x[512,128] ----
    {
        int totalWaves = ((N + 15) / 16) * (128 / 64);
        int blocks = (totalWaves * 32 + 255) / 256;
        wmma_gemm_kernel<<<blocks, 256, 0, stream>>>(x1, W2p, h2, N, 512, 128, totalWaves);
    }

    // ---- conv2 attention (reused buffers need re-zero) ----
    zero_kernel<<<gsgrid((long long)N), 256, 0, stream>>>(mk2, (long long)N);
    zero_kernel<<<gsgrid((long long)N), 256, 0, stream>>>((unsigned*)den2, (long long)N);
    zero_kernel<<<gsgrid((long long)N * 128), 256, 0, stream>>>((unsigned*)out2, (long long)N * 128);

    attn_vec_kernel<<<N, 128, 2 * 128 * 4, stream>>>(h2, att_s2, att_d2, as2, ad2, 128, 1, 128);
    edge_max_kernel<<<gsgrid((long long)Et), 256, 0, stream>>>(src, dst, E, (long long)Et,
                                                               as2, ad2, mk2, 1);
    edge_exp_kernel<<<gsgrid((long long)Et), 256, 0, stream>>>(src, dst, E, (long long)Et,
                                                               as2, ad2, mk2, ex2, den2, 1);
    edge_msg_kernel<<<gsgrid((long long)Et * 128), 256, 0, stream>>>(src, dst, E, (long long)Et * 128,
                                                                     h2, ex2, den2, out2, 1, 128);
    bias_elu_f32_kernel<<<gsgrid((long long)N * 128), 256, 0, stream>>>(out2, b2, x2,
                                                                        (long long)N * 128, 128);

    // ---- pool + classifier ----
    pool_max_kernel<<<gsgrid((long long)N * 128), 256, 0, stream>>>(x2, batch, pk,
                                                                    (long long)N * 128, 128);
    cls1_kernel<<<64, 64, 0, stream>>>(pk, Wc1, bc1, hcls);
    cls2_kernel<<<64, 8, 0, stream>>>(hcls, Wc2, bc2, (float*)d_out);

    (void)n_in; (void)out_size; (void)ws_size;
}
